// MILPGNNModel_80857054314575
// MI455X (gfx1250) — compile-verified, matrix operand order checked by reading, not run on hardware
//
#include <hip/hip_runtime.h>

typedef __attribute__((ext_vector_type(16))) __bf16          v16bf;
typedef __attribute__((ext_vector_type(8)))  float           v8f;

#define H 128
#define LN_EPS 1e-5f
#define SWP 40   // LDS row pitch (shorts) for 32-wide W tile, padded to spread banks

__device__ __forceinline__ __bf16 tobf(float f) {
    return (__bf16)f;   // native f32->bf16 (RNE) -> v_cvt_pk_bf16_f32
}

__device__ __forceinline__ unsigned short f2bf_bits(float f) {
    // RNE fp32 -> bf16 bit pattern (used only in the tiny weight pre-pass)
    unsigned int u = __float_as_uint(f);
    unsigned int r = (u + 0x7FFFu + ((u >> 16) & 1u)) >> 16;
    return (unsigned short)r;
}

__device__ __forceinline__ float sigmoidf(float x) {
    return 1.0f / (1.0f + __expf(-x));
}

// one-time f32 -> bf16 weight conversion (weights total ~192 KB as bf16)
__global__ __launch_bounds__(256)
void cvt_weights(const float* __restrict__ src, unsigned short* __restrict__ dst, int len)
{
    for (int i = blockIdx.x * 256 + threadIdx.x; i < len; i += gridDim.x * 256)
        dst[i] = f2bf_bits(src[i]);
}

// Y(M,128) = relu( LN( [X0/cnt | X1] @ Wb^T + bias ) )   (FUSED, KK=256)
// Y(M,128) =             X0 @ Wb^T + bias                (plain, KK=128)
// Wb is (128, KK) row-major bf16. One wave computes a 16x128 output strip.
template <int KK, bool FUSED>
__global__ __launch_bounds__(256)
void gemm_wmma(const float* __restrict__ X0,
               const float* __restrict__ X1,
               const float* __restrict__ cnt,
               const unsigned short* __restrict__ Wb,
               const float* __restrict__ bias,
               const float* __restrict__ ln_g,
               const float* __restrict__ ln_b,
               float* __restrict__ Y,
               int M)
{
    __shared__ unsigned short sW[H * SWP];

    const int tid  = threadIdx.x;
    const int lane = tid & 31;
    const int wave = tid >> 5;
    const int half = lane >> 4;      // 0 = lanes 0-15, 1 = lanes 16-31
    const int l16  = lane & 15;
    const int rowbase = blockIdx.x * 128 + wave * 16;
    const bool active = (rowbase + 16) <= M;

    // 8 col-tiles of 16, f32 accumulators seeded with bias (same for all rows)
    v8f acc[8];
#pragma unroll
    for (int t = 0; t < 8; ++t) {
        float bv = bias[t * 16 + l16];
#pragma unroll
        for (int j = 0; j < 8; ++j) acc[t][j] = bv;
    }

    const int r = rowbase + l16;     // this lane's A-matrix row
    float rowScale = 1.0f;
    if (FUSED && active) rowScale = 1.0f / fmaxf(cnt[r], 1.0f);   // scatter-mean divide
    const int koff = half * 8;       // K sub-offset per ISA 16-bit A/B lane layout

    for (int kbase = 0; kbase < KK; kbase += 32) {
        // stage W tile (128 x 32 bf16 = 8 KB) into LDS: pure uint copy, no conversion
        for (int idx = tid; idx < H * 16; idx += 256) {       // 16 uints per row of 32 shorts
            int n  = idx >> 4;
            int k2 = idx & 15;
            ((unsigned int*)(sW + n * SWP))[k2] =
                ((const unsigned int*)(Wb + n * KK + kbase))[k2];
        }
        __syncthreads();

        if (active) {
            // A operand: lane l holds row r; elems 0-7 = K koff..+7, 8-15 = K koff+16..+23
            const float* src;
            float sc;
            if (FUSED && kbase >= H) { src = X1 + r * H + (kbase - H); sc = 1.0f; }
            else                     { src = X0 + r * H + kbase;       sc = FUSED ? rowScale : 1.0f; }
            v16bf a;
#pragma unroll
            for (int i = 0; i < 8; ++i) a[i]     = tobf(src[koff + i]      * sc);
#pragma unroll
            for (int i = 0; i < 8; ++i) a[8 + i] = tobf(src[koff + 16 + i] * sc);

            // B operand loader: column n of B = row n of Wb (B = Wb^T), same lane layout
            auto loadB = [&](v16bf& b, int t) {
                const unsigned short* wr = &sW[(t * 16 + l16) * SWP + koff];
#pragma unroll
                for (int i = 0; i < 8; ++i) {
                    b[i]     = __builtin_bit_cast(__bf16, wr[i]);
                    b[8 + i] = __builtin_bit_cast(__bf16, wr[16 + i]);
                }
            };

            // software-pipeline B tiles with two alternating live buffers (no copies,
            // so the compiler cannot coalesce them back into one register octet):
            // ds loads for tile t+1 are in flight while wmma(t) executes.
            v16bf bb[2];
            loadB(bb[0], 0);
#pragma unroll
            for (int t = 0; t < 8; ++t) {
                if (t < 7) loadB(bb[(t + 1) & 1], t + 1);
                acc[t] = __builtin_amdgcn_wmma_f32_16x16x32_bf16(
                    false, a, false, bb[t & 1], (short)0, acc[t], false, false);
            }
        }
        __syncthreads();
    }

    if (!active) return;

    if (FUSED) {
        // fused LayerNorm + ReLU. Row m's 128 values live in acc[0..7][j] across the
        // 16 lanes of this half-wave -> shfl_xor reduction over masks 1,2,4,8.
        float gv[8], bv[8];
#pragma unroll
        for (int t = 0; t < 8; ++t) {
            gv[t] = ln_g[t * 16 + l16];
            bv[t] = ln_b[t * 16 + l16];
        }
#pragma unroll
        for (int j = 0; j < 8; ++j) {
            float s = 0.0f, q = 0.0f;
#pragma unroll
            for (int t = 0; t < 8; ++t) { float v = acc[t][j]; s += v; q += v * v; }
#pragma unroll
            for (int m = 1; m < 16; m <<= 1) {
                s += __shfl_xor(s, m, 32);
                q += __shfl_xor(q, m, 32);
            }
            float mean = s * (1.0f / 128.0f);
            float var  = q * (1.0f / 128.0f) - mean * mean;
            float rstd = rsqrtf(var + LN_EPS);
            int row = rowbase + j + half * 8;    // C/D layout: VGPR j -> M = j (+8 upper half)
            float* yo = Y + row * H;
#pragma unroll
            for (int t = 0; t < 8; ++t) {
                float v = (acc[t][j] - mean) * rstd * gv[t] + bv[t];
                yo[t * 16 + l16] = fmaxf(v, 0.0f);
            }
        }
    } else {
#pragma unroll
        for (int j = 0; j < 8; ++j) {
            int row = rowbase + j + half * 8;
            float* yo = Y + row * H;
#pragma unroll
            for (int t = 0; t < 8; ++t) yo[t * 16 + l16] = acc[t][j];
        }
    }
}

// One wave per edge: gather lin[src], gate by sigmoid(ew*gw+gb), atomic scatter-add
// into agg[dst], count into cnt[dst].
__global__ __launch_bounds__(256)
void edge_scatter(const long long* __restrict__ src_idx,
                  const long long* __restrict__ dst_idx,
                  const float* __restrict__ lin,
                  const float* __restrict__ ew,
                  const float* __restrict__ gw,
                  const float* __restrict__ gb,
                  float* __restrict__ agg,
                  float* __restrict__ cnt,
                  int E)
{
    int eid = blockIdx.x * 8 + (threadIdx.x >> 5);
    if (eid >= E) return;
    int lane = threadIdx.x & 31;

    long long sr = src_idx[eid];
    long long dr = dst_idx[eid];
    float e = ew[eid];

    const float4* row = (const float4*)(lin + sr * H);
    float4 v = row[lane];
    float4 w4 = ((const float4*)gw)[lane];
    float4 b4 = ((const float4*)gb)[lane];

    float g0 = sigmoidf(e * w4.x + b4.x);
    float g1 = sigmoidf(e * w4.y + b4.y);
    float g2 = sigmoidf(e * w4.z + b4.z);
    float g3 = sigmoidf(e * w4.w + b4.w);

    float* dst = agg + dr * H + lane * 4;
    atomicAdd(dst + 0, v.x * g0);
    atomicAdd(dst + 1, v.y * g1);
    atomicAdd(dst + 2, v.z * g2);
    atomicAdd(dst + 3, v.w * g3);
    if (lane == 0) atomicAdd(cnt + dr, 1.0f);
}

extern "C" void kernel_launch(void* const* d_in, const int* in_sizes, int n_in,
                              void* d_out, int out_size, void* d_ws, size_t ws_size,
                              hipStream_t stream)
{
    const float*     vh         = (const float*)d_in[0];
    const float*     ch         = (const float*)d_in[1];
    const long long* edge_index = (const long long*)d_in[2];
    const float*     ew         = (const float*)d_in[3];
    const float*     v2c_lin_w  = (const float*)d_in[4];
    const float*     v2c_lin_b  = (const float*)d_in[5];
    const float*     v2c_gate_w = (const float*)d_in[6];
    const float*     v2c_gate_b = (const float*)d_in[7];
    const float*     v2c_upd_w  = (const float*)d_in[8];
    const float*     v2c_upd_b  = (const float*)d_in[9];
    const float*     v2c_ln_g   = (const float*)d_in[10];
    const float*     v2c_ln_b   = (const float*)d_in[11];
    const float*     c2v_lin_w  = (const float*)d_in[12];
    const float*     c2v_lin_b  = (const float*)d_in[13];
    const float*     c2v_gate_w = (const float*)d_in[14];
    const float*     c2v_gate_b = (const float*)d_in[15];
    const float*     c2v_upd_w  = (const float*)d_in[16];
    const float*     c2v_upd_b  = (const float*)d_in[17];
    const float*     c2v_ln_g   = (const float*)d_in[18];
    const float*     c2v_ln_b   = (const float*)d_in[19];

    const int N = in_sizes[0] / H;     // 200000
    const int C = in_sizes[1] / H;     // 50000
    const int E = in_sizes[3];         // 600000
    const long long* ci = edge_index;        // row 0: constraint idx
    const long long* vi = edge_index + E;    // row 1: variable idx

    // workspace layout:
    //   W0 (N*H f32)  = vh_lin, later agg_v
    //   W1 (C*H f32)  = agg_c, later ch_lin
    //   W2 (N   f32)  = counts
    //   Wb* (bf16)    = pre-converted weights (~192 KB)
    float* W0 = (float*)d_ws;
    float* W1 = W0 + (size_t)N * H;
    float* W2 = W1 + (size_t)C * H;
    unsigned short* Wb_v2c_lin = (unsigned short*)(W2 + N);
    unsigned short* Wb_v2c_upd = Wb_v2c_lin + H * H;
    unsigned short* Wb_c2v_lin = Wb_v2c_upd + H * 2 * H;
    unsigned short* Wb_c2v_upd = Wb_c2v_lin + H * H;

    float* vh_new = (float*)d_out;
    float* ch_new = vh_new + (size_t)N * H;

    dim3 blk(256);
    int gN = (N + 127) / 128;
    int gC = (C + 127) / 128;
    int gE = (E + 7) / 8;

    // weight pre-conversion (tiny)
    cvt_weights<<<64, blk, 0, stream>>>(v2c_lin_w, Wb_v2c_lin, H * H);
    cvt_weights<<<64, blk, 0, stream>>>(v2c_upd_w, Wb_v2c_upd, H * 2 * H);
    cvt_weights<<<64, blk, 0, stream>>>(c2v_lin_w, Wb_c2v_lin, H * H);
    cvt_weights<<<64, blk, 0, stream>>>(c2v_upd_w, Wb_c2v_upd, H * 2 * H);

    // ---- v2c half ----
    hipMemsetAsync(W1, 0, (size_t)C * H * sizeof(float), stream);
    hipMemsetAsync(W2, 0, (size_t)C * sizeof(float), stream);
    gemm_wmma<H, false><<<gN, blk, 0, stream>>>(vh, nullptr, nullptr,
                                                Wb_v2c_lin, v2c_lin_b,
                                                nullptr, nullptr, W0, N);
    edge_scatter<<<gE, blk, 0, stream>>>(vi, ci, W0, ew, v2c_gate_w, v2c_gate_b, W1, W2, E);
    gemm_wmma<2 * H, true><<<gC, blk, 0, stream>>>(W1, ch, W2,
                                                   Wb_v2c_upd, v2c_upd_b,
                                                   v2c_ln_g, v2c_ln_b, ch_new, C);

    // ---- c2v half ----
    hipMemsetAsync(W0, 0, (size_t)N * H * sizeof(float), stream);
    hipMemsetAsync(W2, 0, (size_t)N * sizeof(float), stream);
    gemm_wmma<H, false><<<gC, blk, 0, stream>>>(ch_new, nullptr, nullptr,
                                                Wb_c2v_lin, c2v_lin_b,
                                                nullptr, nullptr, W1, C);
    edge_scatter<<<gE, blk, 0, stream>>>(ci, vi, W1, ew, c2v_gate_w, c2v_gate_b, W0, W2, E);
    gemm_wmma<2 * H, true><<<gN, blk, 0, stream>>>(W0, vh, W2,
                                                   Wb_c2v_upd, c2v_upd_b,
                                                   c2v_ln_g, c2v_ln_b, vh_new, N);

    (void)n_in; (void)out_size; (void)ws_size;
}
